// GConvNetBigGraph_26310969655871
// MI455X (gfx1250) — compile-verified, matrix operand order checked by reading, not run on hardware
//
#include <hip/hip_runtime.h>
#include <hip/hip_bf16.h>

typedef __attribute__((ext_vector_type(16))) _Float16 v16h;
typedef __attribute__((ext_vector_type(8)))  _Float16 v8h;
typedef __attribute__((ext_vector_type(8)))  float    v8f;

#define G_GRAPHS 100
#define K_PAD    32     // K=30 padded to 32
#define NTILES   32     // 500 cols padded to 512 -> 32 tiles of 16
#define SPLIT    5      // blocks per graph in the pool kernel

__device__ __forceinline__ float fast_sigmoid(float x) {
    return __builtin_amdgcn_rcpf(1.0f + __expf(-x));
}

// ---------------- init: zero agg + degree arrays ----------------
__global__ void init_kernel(float* agg, float* degO, float* degI, int N) {
    int i = blockIdx.x * 256 + threadIdx.x;
    if (i < N * K_PAD) agg[i] = 0.0f;
    if (i < N) { degO[i] = 0.0f; degI[i] = 0.0f; }
}

// ---------------- degrees ----------------
__global__ void degree_kernel(const int* __restrict__ src, const int* __restrict__ dst,
                              float* degO, float* degI, int E) {
    int e = blockIdx.x * 256 + threadIdx.x;
    if (e < E) {
        atomicAdd(&degO[src[e]], 1.0f);
        atomicAdd(&degI[dst[e]], 1.0f);
    }
}

// ---------------- invsqrt(max(deg,1)) in place ----------------
__global__ void invdeg_kernel(float* degO, float* degI, int N) {
    int n = blockIdx.x * 256 + threadIdx.x;
    if (n < N) {
        degO[n] = rsqrtf(fmaxf(degO[n], 1.0f));
        degI[n] = rsqrtf(fmaxf(degI[n], 1.0f));
    }
}

// ---------------- edge scatter: agg[dst] += feat[src]*invO[src] ----------------
// one wave per edge, lanes 0..29 cover the 30 features (coalesced 120B row)
__global__ void scatter_kernel(const float* __restrict__ feat,
                               const int* __restrict__ src, const int* __restrict__ dst,
                               const float* __restrict__ invO,
                               float* agg, int E) {
    int t = blockIdx.x * 256 + threadIdx.x;
    int e = t >> 5;
    int j = t & 31;
    if (e >= E) return;
    int s = src[e];
    int d = dst[e];
    if (j < 30) {
        float v = feat[s * 30 + j] * invO[s];
        atomicAdd(&agg[d * K_PAD + j], v);
    }
}

// ---------------- pack W1 [30,500] f32 -> WMMA B-fragment layout (f16) ----------
// B tile j covers cols 16j..16j+15. Lane L holds col N=16j+L%16, half=L/16,
// 16 halves in ISA K-order: elems 0..7 -> K=half*8+i, elems 8..15 -> K=16+half*8+(i-8)
__global__ void packW1_kernel(const float* __restrict__ W1, _Float16* Bpack) {
    int t = blockIdx.x * 256 + threadIdx.x;
    if (t >= NTILES * 32) return;
    int j    = t >> 5;
    int lane = t & 31;
    int half = lane >> 4;
    int col  = lane & 15;
    int n = j * 16 + col;
    _Float16* out = Bpack + t * 16;
#pragma unroll
    for (int i = 0; i < 16; i++) {
        int K = (i < 8) ? (half * 8 + i) : (16 + half * 8 + (i - 8));
        float v = (K < 30 && n < 500) ? W1[K * 500 + n] : 0.0f;
        out[i] = (_Float16)v;
    }
}

// ---------------- fused GEMM (agg*invI @ W1 + b1) -> SiLU -> partial max ------
// SPLIT blocks per graph; 8 waves; wave w owns column tiles {w,w+8,w+16,w+24}.
// Software-pipelined: next A-tile is prefetched into registers during compute.
__global__ void gemm_pool_kernel(const float* __restrict__ agg,
                                 const float* __restrict__ invI,
                                 const _Float16* __restrict__ Bpack,
                                 const float* __restrict__ b1,
                                 float* part, int nodes_per_graph) {
    int g    = blockIdx.x % G_GRAPHS;
    int sp   = blockIdx.x / G_GRAPHS;
    int t    = threadIdx.x;
    int wave = t >> 5;
    int lane = t & 31;
    int half = lane >> 4;
    int col  = lane & 15;

    __shared__ __align__(16) _Float16 As[16 * K_PAD];

    v16h  bfrag[4];
    float bias[4];
    float runmax[4][8];
#pragma unroll
    for (int i = 0; i < 4; i++) {
        int j = wave + i * 8;
        bfrag[i] = *(const v16h*)(Bpack + ((j * 32) + lane) * 16);
        int n = j * 16 + col;
        bias[i] = (n < 500) ? b1[n] : 0.0f;
#pragma unroll
        for (int d = 0; d < 8; d++) runmax[i][d] = -1e30f;
    }

    int mtiles = nodes_per_graph / 16;
    int mt0 = (mtiles * sp) / SPLIT;
    int mt1 = (mtiles * (sp + 1)) / SPLIT;

    int rowsel = t >> 4;        // 0..15
    int csel   = (t & 15) * 2;  // 0,2,..,30
    const float* base = agg + ((size_t)g * nodes_per_graph) * K_PAD;
    const float* ibase = invI + (size_t)g * nodes_per_graph;

    // prefetch first tile
    float  s = ibase[mt0 * 16 + rowsel];
    float2 v = *(const float2*)(base + (size_t)(mt0 * 16 + rowsel) * K_PAD + csel);

    for (int mt = mt0; mt < mt1; mt++) {
        __syncthreads();                          // prev iter done reading As
        As[rowsel * K_PAD + csel]     = (_Float16)(v.x * s);
        As[rowsel * K_PAD + csel + 1] = (_Float16)(v.y * s);
        __syncthreads();                          // As ready

        // prefetch next tile (overlaps the compute below)
        if (mt + 1 < mt1) {
            int node = (mt + 1) * 16 + rowsel;
            s = ibase[node];
            v = *(const float2*)(base + (size_t)node * K_PAD + csel);
        }

        // A fragment: row M=lane%16, K-order per ISA layout (two 16B chunks)
        const _Float16* rowp = As + col * K_PAD;
        v8h lo = *(const v8h*)(rowp + half * 8);
        v8h hi = *(const v8h*)(rowp + 16 + half * 8);
        v16h a;
#pragma unroll
        for (int i = 0; i < 8; i++) { a[i] = lo[i]; a[i + 8] = hi[i]; }

#pragma unroll
        for (int i = 0; i < 4; i++) {
            v8f c = {};
            c = __builtin_amdgcn_wmma_f32_16x16x32_f16(
                    false, a, false, bfrag[i], (short)0, c, false, false);
#pragma unroll
            for (int d = 0; d < 8; d++) {
                float x = c[d] + bias[i];
                runmax[i][d] = fmaxf(runmax[i][d], x * fast_sigmoid(x));
            }
        }
    }

    // reduce 8 accumulator VGPRs (M within tile) + the two lane halves
#pragma unroll
    for (int i = 0; i < 4; i++) {
        float m = runmax[i][0];
#pragma unroll
        for (int d = 1; d < 8; d++) m = fmaxf(m, runmax[i][d]);
        m = fmaxf(m, __shfl_xor(m, 16, 32));
        int n = (wave + i * 8) * 16 + col;
        if (half == 0 && n < 500) part[((size_t)sp * G_GRAPHS + g) * 500 + n] = m;
    }
}

// ---------------- reduce SPLIT partial maxima -> pooled [100,500] -------------
__global__ void reduce_pool_kernel(const float* __restrict__ part, float* pooled) {
    int i = blockIdx.x * 256 + threadIdx.x;
    if (i >= G_GRAPHS * 500) return;
    float m = -1e30f;
#pragma unroll
    for (int sp = 0; sp < SPLIT; sp++)
        m = fmaxf(m, part[(size_t)sp * G_GRAPHS * 500 + i]);
    pooled[i] = m;
}

// ---------------- tail MLP: y=silu(pooled@W2+b2); out=sigmoid(y@W3+b3) --------
__global__ void mlp_kernel(const float* __restrict__ pooled,
                           const float* __restrict__ W2, const float* __restrict__ b2,
                           const float* __restrict__ W3, const float* __restrict__ b3,
                           float* out) {
    int g = blockIdx.x;
    int l = threadIdx.x;
    __shared__ float ysh[20];
    if (l < 20) {
        float acc = b2[l];
        for (int k = 0; k < 500; k++) acc += pooled[g * 500 + k] * W2[k * 20 + l];
        ysh[l] = acc * fast_sigmoid(acc);
    }
    __syncthreads();
    if (l < 4) {
        float acc = b3[l];
#pragma unroll
        for (int j = 0; j < 20; j++) acc += ysh[j] * W3[j * 4 + l];
        out[g * 4 + l] = fast_sigmoid(acc);
    }
}

extern "C" void kernel_launch(void* const* d_in, const int* in_sizes, int n_in,
                              void* d_out, int out_size, void* d_ws, size_t ws_size,
                              hipStream_t stream) {
    const float* feat = (const float*)d_in[0];
    const int*   src  = (const int*)d_in[1];
    const int*   dst  = (const int*)d_in[2];
    const float* W1   = (const float*)d_in[5];
    const float* b1   = (const float*)d_in[6];
    const float* W2   = (const float*)d_in[7];
    const float* b2   = (const float*)d_in[8];
    const float* W3   = (const float*)d_in[9];
    const float* b3   = (const float*)d_in[10];
    float* out = (float*)d_out;

    int N = in_sizes[0] / 30;
    int E = in_sizes[1];
    int npg = N / G_GRAPHS;

    // workspace layout (all sections 32B aligned)
    float* agg    = (float*)d_ws;                     // N * 32
    float* degO   = agg + (size_t)N * K_PAD;          // N
    float* degI   = degO + N;                         // N
    float* part   = degI + N;                         // SPLIT * 100 * 500
    float* pooled = part + (size_t)SPLIT * G_GRAPHS * 500;  // 100 * 500
    _Float16* Bpack = (_Float16*)(pooled + G_GRAPHS * 500); // 32*32*16 halves

    init_kernel<<<(N * K_PAD + 255) / 256, 256, 0, stream>>>(agg, degO, degI, N);
    degree_kernel<<<(E + 255) / 256, 256, 0, stream>>>(src, dst, degO, degI, E);
    invdeg_kernel<<<(N + 255) / 256, 256, 0, stream>>>(degO, degI, N);
    packW1_kernel<<<(NTILES * 32 + 255) / 256, 256, 0, stream>>>(W1, Bpack);
    scatter_kernel<<<(E * 32 + 255) / 256, 256, 0, stream>>>(feat, src, dst, degO, agg, E);
    gemm_pool_kernel<<<G_GRAPHS * SPLIT, 256, 0, stream>>>(agg, degI, Bpack, b1, part, npg);
    reduce_pool_kernel<<<(G_GRAPHS * 500 + 255) / 256, 256, 0, stream>>>(part, pooled);
    mlp_kernel<<<G_GRAPHS, 32, 0, stream>>>(pooled, W2, b2, W3, b3, out);
}
